// BahdanauAttention_34102040330496
// MI455X (gfx1250) — compile-verified, matrix operand order checked by reading, not run on hardware
//
#include <hip/hip_runtime.h>
#include <hip/hip_bf16.h>

// ---------- problem constants ----------
#define B_   32
#define M_   2048
#define QS_  1024
#define KS_  1024
#define H_   1024
#define CS_  128

// ---------- WMMA types / helpers (gfx1250, wave32) ----------
typedef __bf16 v16bf __attribute__((ext_vector_type(16)));
typedef float  v8f   __attribute__((ext_vector_type(8)));
typedef int    v4i_  __attribute__((ext_vector_type(4)));

// D = A(16x32 bf16) * B(32x16 bf16) + C(16x16 f32)
__device__ __forceinline__ v8f wmma_bf16(v16bf a, v16bf b, v8f c) {
    return __builtin_amdgcn_wmma_f32_16x16x32_bf16(
        /*neg_a=*/false, a, /*neg_b=*/false, b,
        /*c_mod=*/(short)0, c, /*reuse_a=*/false, /*reuse_b=*/false);
}

// Per-lane fragment: lane holds row (lane&15); K chunks [koff..koff+7] and
// [koff+16..koff+23] where koff = (lane<16)?0:8  (ISA 7.12.2 16-bit 16x32 layout).
__device__ __forceinline__ v16bf frag_bf(const __bf16* p, int koff) {
    v16bf f;
#pragma unroll
    for (int i = 0; i < 8; ++i) f[i] = p[koff + i];
#pragma unroll
    for (int i = 0; i < 8; ++i) f[8 + i] = p[koff + 16 + i];
    return f;
}

// Same fragment but from an f32 source (converts on load).
__device__ __forceinline__ v16bf frag_f32(const float* p, int koff) {
    v16bf f;
#pragma unroll
    for (int i = 0; i < 8; ++i) f[i] = (__bf16)p[koff + i];
#pragma unroll
    for (int i = 0; i < 8; ++i) f[8 + i] = (__bf16)p[koff + 16 + i];
    return f;
}

__device__ __forceinline__ float sigmoidf_(float x) { return 1.0f / (1.0f + __expf(-x)); }

// ---------- async copy (gfx1250 GLOBAL_LOAD_ASYNC_TO_LDS_B128), guarded ----------
#if __has_builtin(__builtin_amdgcn_global_load_async_to_lds_b128)
#define ASYNC_LDS 1
#else
#define ASYNC_LDS 0
#endif

__device__ __forceinline__ void copy16_to_lds(const float* g, float* l) {
#if ASYNC_LDS
    __builtin_amdgcn_global_load_async_to_lds_b128(
        (__attribute__((address_space(1))) v4i_*)(g),
        (__attribute__((address_space(3))) v4i_*)(l), 0, 0);
#else
    *(float4*)l = *(const float4*)g;
#endif
}
__device__ __forceinline__ void async_copies_wait() {
#if ASYNC_LDS
#if __has_builtin(__builtin_amdgcn_s_wait_asynccnt)
    __builtin_amdgcn_s_wait_asynccnt(0);
#else
    asm volatile("s_wait_asynccnt 0" ::: "memory");
#endif
#endif
}

// ---------- vectorized f32 -> bf16 LDS tile staging ----------
// dst rows padded to COLS+8 elems (row stride 16B-aligned when COLS%8==0).
template <int ROWS, int COLS>
__device__ __forceinline__ void stage_bf16(const float* __restrict__ src, int srcStride,
                                           __bf16 (*dst)[COLS + 8], int tid) {
    const int TOT = ROWS * COLS / 8;
    for (int i = tid; i < TOT; i += 256) {
        const int e = i * 8;
        const int r = e / COLS, c = e % COLS;
        const float4* s = (const float4*)(src + (size_t)r * srcStride + c);
        float4 a = s[0], b = s[1];
        __bf16 tmp[8] __attribute__((aligned(16))) = {
            (__bf16)a.x, (__bf16)a.y, (__bf16)a.z, (__bf16)a.w,
            (__bf16)b.x, (__bf16)b.y, (__bf16)b.z, (__bf16)b.w};
        *(uint4*)(&dst[r][c]) = *(const uint4*)tmp;
    }
}

// ---------- weight f32 -> bf16 conversion (once per launch, into ws) ----------
__global__ __launch_bounds__(256) void cvt_bf16_kernel(const float* __restrict__ src,
                                                       __bf16* __restrict__ dst, int n) {
    int i = blockIdx.x * 256 + threadIdx.x;
    if (i < n) dst[i] = (__bf16)src[i];
}

// ---------- kernel P: small query projections + bias folding ----------
__global__ __launch_bounds__(256) void proj_kernel(
    const float* __restrict__ query,
    const float* __restrict__ Wq,  const float* __restrict__ bq,
    const float* __restrict__ Wqc, const float* __restrict__ bqc,
    const float* __restrict__ bk,  const float* __restrict__ bc,
    const float* __restrict__ bkc, const float* __restrict__ bcc,
    const float* __restrict__ bac,
    float* __restrict__ scorebase, float* __restrict__ cubase)
{
    __shared__ float qrow[QS_];
    const int idx = blockIdx.x * 256 + threadIdx.x;   // b*H + h, b constant per block
    const int b = idx / H_;
    const int h = idx % H_;
    const float* qp = query + (size_t)b * QS_;
    for (int i = threadIdx.x; i < QS_; i += 256) qrow[i] = qp[i];
    __syncthreads();
    const float* wq  = Wq  + (size_t)h * QS_;
    const float* wqc = Wqc + (size_t)h * QS_;
    float s0 = 0.f, s1 = 0.f;
    for (int k = 0; k < QS_; ++k) {
        s0 = fmaf(qrow[k], wq[k],  s0);
        s1 = fmaf(qrow[k], wqc[k], s1);
    }
    scorebase[idx] = s0 + bq[h]  + bk[h]  + bc[h];
    cubase[idx]    = s1 + bqc[h] + bkc[h] + bcc[h] + bac[h];
}

// ---------- kernel A: attention scores ----------
// scores[b,m] = Wl . tanh(scorebase[b,:] + memory[b,m]@Wk^T + coverage[b,m]@Wc^T) + bl
__global__ __launch_bounds__(256) void scores_kernel(
    const float* __restrict__ memory, const float* __restrict__ coverage,
    const __bf16* __restrict__ bWk, const __bf16* __restrict__ bWc,
    const float* __restrict__ Wl, const float* __restrict__ bl,
    const float* __restrict__ scorebase, float* __restrict__ scores_out)
{
    __shared__ __bf16 ldsA[16][KS_ + 8] __attribute__((aligned(16))); // memory tile, bf16
    __shared__ float  covF[16 * CS_];                                  // coverage tile, f32 (async)
    __shared__ float  sScore[16];

    const int tid    = threadIdx.x;
    const int lane   = tid & 31;
    const int wave   = tid >> 5;
    const int lane16 = lane & 15;
    const int half   = lane >> 4;
    const int koff   = half ? 8 : 0;
    const int row0   = blockIdx.x * 16;
    const int b      = row0 / M_;
    const int m0     = row0 % M_;

    const float* memTile = memory   + ((size_t)b * M_ + m0) * KS_;
    const float* covTile = coverage + ((size_t)b * M_ + m0) * CS_;  // contiguous 16*128 floats

    // async-copy coverage tile (8KB) raw f32 into LDS; 2 x 16B per thread
    copy16_to_lds(covTile + tid * 4,        covF + tid * 4);
    copy16_to_lds(covTile + tid * 4 + 1024, covF + tid * 4 + 1024);

    stage_bf16<16, KS_>(memTile, KS_, ldsA, tid);
    if (tid < 16) sScore[tid] = 0.f;
    async_copies_wait();
    __syncthreads();

    float part[8];
#pragma unroll
    for (int i = 0; i < 8; ++i) part[i] = 0.f;

    for (int ht = 0; ht < 8; ++ht) {
        const int h0 = (wave * 8 + ht) * 16;
        v8f acc = {};
        const __bf16* wkRow = bWk + (size_t)(h0 + lane16) * KS_;
        for (int kk = 0; kk < KS_ / 32; ++kk) {
            v16bf a  = frag_bf(&ldsA[lane16][kk * 32], koff);
            v16bf bb = frag_bf(wkRow + kk * 32, koff);
            acc = wmma_bf16(a, bb, acc);
        }
        const __bf16* wcRow = bWc + (size_t)(h0 + lane16) * CS_;
        for (int kk = 0; kk < CS_ / 32; ++kk) {
            v16bf a  = frag_f32(&covF[lane16 * CS_ + kk * 32], koff);
            v16bf bb = frag_bf(wcRow + kk * 32, koff);
            acc = wmma_bf16(a, bb, acc);
        }
        const int   h  = h0 + lane16;
        const float sb = scorebase[b * H_ + h];
        const float wl = Wl[h];
#pragma unroll
        for (int i = 0; i < 8; ++i)
            part[i] = fmaf(tanhf(acc[i] + sb), wl, part[i]);
    }

    // cross-lane reduce over the 16 N-lanes; m = i + 8*half
#pragma unroll
    for (int i = 0; i < 8; ++i) {
        float v = part[i];
        for (int off = 8; off; off >>= 1) v += __shfl_xor(v, off, 16);
        if (lane16 == 0) atomicAdd(&sScore[i + 8 * half], v);
    }
    __syncthreads();
    if (tid < 16)
        scores_out[(size_t)b * M_ + m0 + tid] = sScore[tid] + bl[0];
}

// ---------- kernel B: masked softmax over M (in-place in weights region) ----------
__global__ __launch_bounds__(256) void softmax_kernel(
    float* __restrict__ w, const unsigned char* __restrict__ mask)
{
    __shared__ float red[8];
    const int b = blockIdx.x, tid = threadIdx.x;
    const int lane = tid & 31, wave = tid >> 5;
    float vals[8];
    float mx = -1e30f;
#pragma unroll
    for (int j = 0; j < 8; ++j) {
        int idx = tid + j * 256;
        float s = w[(size_t)b * M_ + idx];
        if (mask[(size_t)b * M_ + idx]) s = -1e30f;
        vals[j] = s;
        mx = fmaxf(mx, s);
    }
    for (int off = 16; off; off >>= 1) mx = fmaxf(mx, __shfl_xor(mx, off, 32));
    if (lane == 0) red[wave] = mx;
    __syncthreads();
    if (tid == 0) { float m = red[0]; for (int i = 1; i < 8; ++i) m = fmaxf(m, red[i]); red[0] = m; }
    __syncthreads();
    mx = red[0];
    __syncthreads();
    float sum = 0.f;
#pragma unroll
    for (int j = 0; j < 8; ++j) { vals[j] = __expf(vals[j] - mx); sum += vals[j]; }
    for (int off = 16; off; off >>= 1) sum += __shfl_xor(sum, off, 32);
    if (lane == 0) red[wave] = sum;
    __syncthreads();
    if (tid == 0) { float s = 0.f; for (int i = 0; i < 8; ++i) s += red[i]; red[0] = s; }
    __syncthreads();
    const float inv = 1.0f / red[0];
#pragma unroll
    for (int j = 0; j < 8; ++j) w[(size_t)b * M_ + tid + j * 256] = vals[j] * inv;
}

// ---------- zero fill ----------
__global__ void zero_kernel(float* __restrict__ p, int n) {
    int i = blockIdx.x * 256 + threadIdx.x;
    if (i < n) p[i] = 0.f;
}

// ---------- kernel C: fused coverage-GRU update + attns ----------
__global__ __launch_bounds__(256) void coverage_kernel(
    const float* __restrict__ memory, const float* __restrict__ coverage,
    const __bf16* __restrict__ bWkc, const __bf16* __restrict__ bWcc,
    const float* __restrict__ Wac, const float* __restrict__ cubase,
    const __bf16* __restrict__ bWih, const float* __restrict__ gru_bih,
    const __bf16* __restrict__ bWhh, const float* __restrict__ gru_bhh,
    const float* __restrict__ weights,
    float* __restrict__ attns, float* __restrict__ newcov)
{
    __shared__ __bf16 cuT[16][H_ + 8] __attribute__((aligned(16))); // cu tile (bf16)
    __shared__ float  covF[16 * CS_];                                // coverage tile, f32 (async)
    __shared__ float  sW[16];                                        // attention weights

    const int tid    = threadIdx.x;
    const int lane   = tid & 31;
    const int wave   = tid >> 5;
    const int lane16 = lane & 15;
    const int half   = lane >> 4;
    const int koff   = half ? 8 : 0;
    const int row0   = blockIdx.x * 16;
    const int b      = row0 / M_;
    const int m0     = row0 % M_;

    const float* covTile = coverage + ((size_t)b * M_ + m0) * CS_;
    copy16_to_lds(covTile + tid * 4,        covF + tid * 4);
    copy16_to_lds(covTile + tid * 4 + 1024, covF + tid * 4 + 1024);
    if (tid < 16) sW[tid] = weights[(size_t)b * M_ + m0 + tid];
    async_copies_wait();
    __syncthreads();

    // ---- phase 1: cu[m,h] = cubase + memory@Wkc^T + coverage@Wcc^T + w[m]*Wac[h] ----
    const float* aRow = memory + ((size_t)b * M_ + m0 + lane16) * KS_;
    for (int ht = 0; ht < 8; ++ht) {
        const int h0 = (wave * 8 + ht) * 16;
        v8f acc = {};
        const __bf16* bRow = bWkc + (size_t)(h0 + lane16) * KS_;
        for (int kk = 0; kk < KS_ / 32; ++kk) {
            v16bf a  = frag_f32(aRow + kk * 32, koff);
            v16bf bb = frag_bf(bRow + kk * 32, koff);
            acc = wmma_bf16(a, bb, acc);
        }
        const __bf16* cRow = bWcc + (size_t)(h0 + lane16) * CS_;
        for (int kk = 0; kk < CS_ / 32; ++kk) {
            v16bf a  = frag_f32(&covF[lane16 * CS_ + kk * 32], koff);
            v16bf bb = frag_bf(cRow + kk * 32, koff);
            acc = wmma_bf16(a, bb, acc);
        }
        const int   h    = h0 + lane16;
        const float base = cubase[b * H_ + h];
        const float wac  = Wac[h];
#pragma unroll
        for (int i = 0; i < 8; ++i) {
            int m = i + 8 * half;
            cuT[m][h] = (__bf16)(acc[i] + base + sW[m] * wac);
        }
    }
    __syncthreads();

    // ---- phase 2: GRU gates; wave 'wave' owns C-columns [c0, c0+16) ----
    const int c0 = wave * 16;
    v8f axr = {}, axz = {}, axn = {}, ahr = {}, ahz = {}, ahn = {};
    const __bf16* wr = bWih + (size_t)(c0 + lane16) * H_;
    const __bf16* wz = bWih + (size_t)(CS_ + c0 + lane16) * H_;
    const __bf16* wn = bWih + (size_t)(2 * CS_ + c0 + lane16) * H_;
    for (int kk = 0; kk < H_ / 32; ++kk) {
        v16bf a = frag_bf(&cuT[lane16][kk * 32], koff);
        axr = wmma_bf16(a, frag_bf(wr + kk * 32, koff), axr);
        axz = wmma_bf16(a, frag_bf(wz + kk * 32, koff), axz);
        axn = wmma_bf16(a, frag_bf(wn + kk * 32, koff), axn);
    }
    const __bf16* hr = bWhh + (size_t)(c0 + lane16) * CS_;
    const __bf16* hz = bWhh + (size_t)(CS_ + c0 + lane16) * CS_;
    const __bf16* hn = bWhh + (size_t)(2 * CS_ + c0 + lane16) * CS_;
    for (int kk = 0; kk < CS_ / 32; ++kk) {
        v16bf a = frag_f32(&covF[lane16 * CS_ + kk * 32], koff);
        ahr = wmma_bf16(a, frag_bf(hr + kk * 32, koff), ahr);
        ahz = wmma_bf16(a, frag_bf(hz + kk * 32, koff), ahz);
        ahn = wmma_bf16(a, frag_bf(hn + kk * 32, koff), ahn);
    }
    const int   c   = c0 + lane16;
    const float bxr = gru_bih[c],           bhr = gru_bhh[c];
    const float bxz = gru_bih[CS_ + c],     bhz = gru_bhh[CS_ + c];
    const float bxn = gru_bih[2 * CS_ + c], bhn = gru_bhh[2 * CS_ + c];
#pragma unroll
    for (int i = 0; i < 8; ++i) {
        const int m = i + 8 * half;
        float r = sigmoidf_(axr[i] + bxr + ahr[i] + bhr);
        float z = sigmoidf_(axz[i] + bxz + ahz[i] + bhz);
        float n = tanhf(axn[i] + bxn + r * (ahn[i] + bhn));
        newcov[((size_t)b * M_ + m0 + m) * CS_ + c] =
            (1.0f - z) * n + z * covF[m * CS_ + c];
    }

    // ---- phase 3: attns[b,k] += sum_m w[m] * memory[b,m,k] ----
    const float* memTile = memory + ((size_t)b * M_ + m0) * KS_;
    for (int col = tid; col < KS_; col += 256) {
        float acc = 0.f;
#pragma unroll 4
        for (int r = 0; r < 16; ++r)
            acc = fmaf(sW[r], memTile[(size_t)r * KS_ + col], acc);
        atomicAdd(&attns[(size_t)b * KS_ + col], acc);
    }
}

// ---------- host launcher ----------
extern "C" void kernel_launch(void* const* d_in, const int* in_sizes, int n_in,
                              void* d_out, int out_size, void* d_ws, size_t ws_size,
                              hipStream_t stream) {
    (void)in_sizes; (void)n_in; (void)out_size; (void)ws_size;

    const float* query    = (const float*)d_in[0];
    const float* memory   = (const float*)d_in[1];
    const float* coverage = (const float*)d_in[2];
    const float* Wq  = (const float*)d_in[3];
    const float* bq  = (const float*)d_in[4];
    const float* Wk  = (const float*)d_in[5];
    const float* bk  = (const float*)d_in[6];
    const float* Wc  = (const float*)d_in[7];
    const float* bc  = (const float*)d_in[8];
    const float* Wcc = (const float*)d_in[9];
    const float* bcc = (const float*)d_in[10];
    const float* Wqc = (const float*)d_in[11];
    const float* bqc = (const float*)d_in[12];
    const float* Wkc = (const float*)d_in[13];
    const float* bkc = (const float*)d_in[14];
    const float* Wac = (const float*)d_in[15];
    const float* bac = (const float*)d_in[16];
    const float* Wl  = (const float*)d_in[17];
    const float* bl  = (const float*)d_in[18];
    const float* gru_wih = (const float*)d_in[19];
    const float* gru_bih = (const float*)d_in[20];
    const float* gru_whh = (const float*)d_in[21];
    const float* gru_bhh = (const float*)d_in[22];
    const unsigned char* mask = (const unsigned char*)d_in[23];

    float* attns   = (float*)d_out;                 // [B, KS]
    float* weights = attns + (size_t)B_ * KS_;      // [B, M]
    float* newcov  = weights + (size_t)B_ * M_;     // [B, M, CS]

    float* scorebase = (float*)d_ws;                // [B, H]
    float* cubase    = scorebase + (size_t)B_ * H_; // [B, H]

    // bf16 copies of the shared weight matrices (L2-resident; read by all WGs)
    __bf16* bWk  = (__bf16*)(cubase + (size_t)B_ * H_);
    __bf16* bWc  = bWk  + (size_t)H_ * KS_;        // 1M
    __bf16* bWkc = bWc  + (size_t)H_ * CS_;        // 128K
    __bf16* bWcc = bWkc + (size_t)H_ * KS_;        // 1M
    __bf16* bWih = bWcc + (size_t)H_ * CS_;        // 128K
    __bf16* bWhh = bWih + (size_t)3 * CS_ * H_;    // 384K

    const int nWk  = H_ * KS_;
    const int nWc  = H_ * CS_;
    const int nWih = 3 * CS_ * H_;
    const int nWhh = 3 * CS_ * CS_;
    cvt_bf16_kernel<<<(nWk  + 255) / 256, 256, 0, stream>>>(Wk,      bWk,  nWk);
    cvt_bf16_kernel<<<(nWc  + 255) / 256, 256, 0, stream>>>(Wc,      bWc,  nWc);
    cvt_bf16_kernel<<<(nWk  + 255) / 256, 256, 0, stream>>>(Wkc,     bWkc, nWk);
    cvt_bf16_kernel<<<(nWc  + 255) / 256, 256, 0, stream>>>(Wcc,     bWcc, nWc);
    cvt_bf16_kernel<<<(nWih + 255) / 256, 256, 0, stream>>>(gru_wih, bWih, nWih);
    cvt_bf16_kernel<<<(nWhh + 255) / 256, 256, 0, stream>>>(gru_whh, bWhh, nWhh);

    // 1) query projections + bias folding
    proj_kernel<<<(B_ * H_) / 256, 256, 0, stream>>>(
        query, Wq, bq, Wqc, bqc, bk, bc, bkc, bcc, bac, scorebase, cubase);

    // 2) attention scores (into weights region)
    scores_kernel<<<(B_ * M_) / 16, 256, 0, stream>>>(
        memory, coverage, bWk, bWc, Wl, bl, scorebase, weights);

    // 3) masked softmax in place
    softmax_kernel<<<B_, 256, 0, stream>>>(weights, mask);

    // 4) zero attns accumulator
    zero_kernel<<<(B_ * KS_ + 255) / 256, 256, 0, stream>>>(attns, B_ * KS_);

    // 5) fused cu-build + GRU + attns accumulation
    coverage_kernel<<<(B_ * M_) / 16, 256, 0, stream>>>(
        memory, coverage, bWkc, bWcc, Wac, cubase,
        bWih, gru_bih, bWhh, gru_bhh, weights, attns, newcov);
}
